// QuantizedLinear_89378269429978
// MI455X (gfx1250) — compile-verified, matrix operand order checked by reading, not run on hardware
//
#include <hip/hip_runtime.h>

// CDNA5 / gfx1250: wave32, WMMA 16x16x32 bf16 -> f32
typedef __attribute__((ext_vector_type(16))) __bf16 bf16x16;
typedef __attribute__((ext_vector_type(2)))  __bf16 bf16x2;
typedef __attribute__((ext_vector_type(8)))  float  f32x8;
typedef __attribute__((ext_vector_type(2)))  float  f32x2;

#define BM 128
#define BN 128
#define BK 32
#define LDT 40          // BK + 8 pad (bf16 elems) -> 80B row pitch, 16B aligned
#define NTHREADS 256    // 8 wave32 waves

__device__ const float NF4_TBL[16] = {
    -1.0f, -0.6962f, -0.5251f, -0.3949f, -0.2844f, -0.1848f, -0.0911f, 0.0f,
     0.0796f, 0.1609f, 0.2461f, 0.3379f, 0.4407f, 0.5626f, 0.723f, 1.0f};

// fp32 pair -> packed bf16 pair; single v_cvt_pk_bf16_f32 (RNE)
__device__ __forceinline__ unsigned f2bf_pack(float a, float b) {
    f32x2 f = {a, b};
    bf16x2 h = __builtin_convertvector(f, bf16x2);
    union { bf16x2 h; unsigned u; } v;
    v.h = h;
    return v.u;
}

__device__ __forceinline__ unsigned short f2bf1(float a) {
    union { __bf16 h; unsigned short u; } v;
    v.h = (__bf16)a;
    return v.u;
}

__global__ __launch_bounds__(NTHREADS)
void nf4_gemm_bf16_wmma(const float* __restrict__ x,      // (M, K) fp32
                        const int*   __restrict__ qw,     // (K, N) codes in [-8,7]
                        const float* __restrict__ scale,  // (K/128 + 1, N) fp32
                        const float* __restrict__ bias,   // (N,) fp32
                        unsigned short* __restrict__ out, // (M, N) bf16 bits
                        int M, int N, int K)
{
    __shared__ __align__(16) unsigned short sA[2][BM * LDT]; // bf16, M-major, K contig
    __shared__ __align__(16) unsigned short sB[2][BN * LDT]; // bf16, N-major, K contig
    __shared__ float tblS[16];

    const int tid  = threadIdx.x;
    const int lane = tid & 31;
    const int wave = tid >> 5;

    if (tid < 16) tblS[tid] = NF4_TBL[tid];

    const int m0 = blockIdx.y * BM;
    const int n0 = blockIdx.x * BN;

    // wave -> 64x32 output patch (2 rows x 4 cols of waves)
    const int wm   = (wave >> 2) * 64;
    const int wn   = (wave & 3) * 32;
    const int lrow = lane & 15;
    const int kk   = (lane >> 4) * 8;    // ISA 16-bit A/B fragment layout

    // B staging coords: fixed column, 16 contiguous k per thread
    const int bn = tid & 127;
    const int bk = (tid >> 7) * 16;

    f32x8 acc[4][2];
    {
        const f32x8 z = {0.f, 0.f, 0.f, 0.f, 0.f, 0.f, 0.f, 0.f};
        #pragma unroll
        for (int i = 0; i < 4; ++i)
            #pragma unroll
            for (int j = 0; j < 2; ++j) acc[i][j] = z;
    }

    float4 areg[4];
    int    breg[16];
    float  sreg;

    // global -> registers for k-tile at k0 (issued early to overlap with WMMA)
    auto load_stage = [&](int k0) {
        #pragma unroll
        for (int i = 0; i < 4; ++i) {
            const int f4 = i * NTHREADS + tid;
            const int r  = f4 >> 3;            // 8 float4 per 32-wide row
            const int c  = (f4 & 7) * 4;
            areg[i] = *reinterpret_cast<const float4*>(
                x + (size_t)(m0 + r) * K + k0 + c);
        }
        sreg = scale[(size_t)(k0 >> 7) * N + n0 + bn];  // BK=32 never crosses group
        #pragma unroll
        for (int i = 0; i < 16; ++i)
            breg[i] = qw[(size_t)(k0 + bk + i) * N + n0 + bn];
    };

    // registers -> LDS (fp32->bf16 cvt for A, NF4 dequant + transpose for B)
    auto store_stage = [&](int buf) {
        #pragma unroll
        for (int i = 0; i < 4; ++i) {
            const int f4 = i * NTHREADS + tid;
            const int r  = f4 >> 3;
            const int c  = (f4 & 7) * 4;
            uint2 v;
            v.x = f2bf_pack(areg[i].x, areg[i].y);
            v.y = f2bf_pack(areg[i].z, areg[i].w);
            *reinterpret_cast<uint2*>(&sA[buf][r * LDT + c]) = v;
        }
        #pragma unroll
        for (int j = 0; j < 4; ++j) {
            const float w0 = tblS[breg[4 * j + 0] + 8] * sreg;
            const float w1 = tblS[breg[4 * j + 1] + 8] * sreg;
            const float w2 = tblS[breg[4 * j + 2] + 8] * sreg;
            const float w3 = tblS[breg[4 * j + 3] + 8] * sreg;
            uint2 v;
            v.x = f2bf_pack(w0, w1);
            v.y = f2bf_pack(w2, w3);
            *reinterpret_cast<uint2*>(&sB[buf][bn * LDT + bk + 4 * j]) = v;
        }
    };

    // 8 WMMAs per k-tile per wave; fragments per ISA 16-bit layout:
    // lanes 0-15: K = kk..kk+7 and kk+16..kk+23 (two ds_load_b128, +0 / +32B)
    auto compute = [&](int buf) {
        bf16x16 afrag[4], bfrag[2];
        #pragma unroll
        for (int mt = 0; mt < 4; ++mt) {
            const unsigned short* p = &sA[buf][(wm + mt * 16 + lrow) * LDT + kk];
            union { uint4 q[2]; bf16x16 v; } u;
            u.q[0] = *reinterpret_cast<const uint4*>(p);
            u.q[1] = *reinterpret_cast<const uint4*>(p + 16);
            afrag[mt] = u.v;
        }
        #pragma unroll
        for (int nt = 0; nt < 2; ++nt) {
            const unsigned short* p = &sB[buf][(wn + nt * 16 + lrow) * LDT + kk];
            union { uint4 q[2]; bf16x16 v; } u;
            u.q[0] = *reinterpret_cast<const uint4*>(p);
            u.q[1] = *reinterpret_cast<const uint4*>(p + 16);
            bfrag[nt] = u.v;
        }
        #pragma unroll
        for (int mt = 0; mt < 4; ++mt)
            #pragma unroll
            for (int nt = 0; nt < 2; ++nt)
                acc[mt][nt] = __builtin_amdgcn_wmma_f32_16x16x32_bf16(
                    false, afrag[mt], false, bfrag[nt],
                    (short)0, acc[mt][nt], false, false);
    };

    const int KT = K / BK;

    // prologue
    load_stage(0);
    __syncthreads();            // tblS visible before dequant
    store_stage(0);
    __syncthreads();

    // software pipeline: loads for t+1 in flight while WMMAs for t execute
    for (int t = 0; t < KT; ++t) {
        const int cur = t & 1;
        if (t + 1 < KT) load_stage((t + 1) * BK);
        compute(cur);
        if (t + 1 < KT) store_stage(cur ^ 1);
        __syncthreads();
    }

    // epilogue: +bias (rounded to bf16 like the reference), RNE bf16 store.
    // C/D layout: lane = {n = lane&15}, rows m = (lane>>4)*8 + r for VGPR r.
    #pragma unroll
    for (int mt = 0; mt < 4; ++mt) {
        #pragma unroll
        for (int nt = 0; nt < 2; ++nt) {
            const int nCol = n0 + wn + nt * 16 + lrow;
            const float bv = (float)(__bf16)bias[nCol];
            #pragma unroll
            for (int r = 0; r < 8; ++r) {
                const int m = m0 + wm + mt * 16 + (lane >> 4) * 8 + r;
                out[(size_t)m * N + nCol] = f2bf1(acc[mt][nt][r] + bv);
            }
        }
    }
}

extern "C" void kernel_launch(void* const* d_in, const int* in_sizes, int n_in,
                              void* d_out, int out_size, void* d_ws, size_t ws_size,
                              hipStream_t stream) {
    const float* x     = (const float*)d_in[0];
    const int*   qw    = (const int*)d_in[1];
    const float* scale = (const float*)d_in[2];
    const float* bias  = (const float*)d_in[3];
    unsigned short* out = (unsigned short*)d_out;

    const int N = in_sizes[3];        // D_OUT = 11008 (bias length)
    const int K = in_sizes[1] / N;    // 4096
    const int M = in_sizes[0] / K;    // 8192

    dim3 grid(N / BN, M / BM);        // 86 x 64
    nf4_gemm_bf16_wmma<<<grid, NTHREADS, 0, stream>>>(x, qw, scale, bias, out, M, N, K);
}
